// MCNWMConv_4612794876596
// MI455X (gfx1250) — matrix operand order repaired and checked by reading, not compiled, not run on hardware
//
#include <hip/hip_runtime.h>
#include <hip/hip_bf16.h>

// Problem constants (from reference)
#define NN   50000
#define EE   800000
#define DD   64
#define CC   3
#define HH   8
#define OUTD 128
#define FEAT 192          // C*D
#define NEG_SLOPE 0.01f

typedef __attribute__((ext_vector_type(16))) _Float16 v16h;
typedef __attribute__((ext_vector_type(8)))  _Float16 h8;
typedef __attribute__((ext_vector_type(8)))  float    v8f;

// ---------------------------------------------------------------------------
// Kernel 1: per-edge per-channel scalar weight from tiny MLP (A=1 -> H=8 -> 1)
// w[c] = elu( sum_i leaky(a*w1[c,i] + b1[c,i]) * w2[c,i] + b2[c] )
// ---------------------------------------------------------------------------
__global__ __launch_bounds__(256) void edgew_kernel(
    const float* __restrict__ ea,   // [E]  (A==1)
    const float* __restrict__ w1,   // [C,1,H]
    const float* __restrict__ b1,   // [C,H]
    const float* __restrict__ w2,   // [C,H,1]
    const float* __restrict__ b2,   // [C,1]
    float* __restrict__ wE)         // [E,4] (4th lane unused, for alignment)
{
    int e = blockIdx.x * 256 + threadIdx.x;
    if (e >= EE) return;
    float a = ea[e];
#pragma unroll
    for (int c = 0; c < CC; ++c) {
        float s = b2[c];
#pragma unroll
        for (int i = 0; i < HH; ++i) {
            float h = a * w1[c * HH + i] + b1[c * HH + i];
            h = (h >= 0.0f) ? h : NEG_SLOPE * h;
            s += h * w2[c * HH + i];
        }
        float w = (s > 0.0f) ? s : (__expf(s) - 1.0f);   // ELU
        wE[(size_t)e * 4 + c] = w;
    }
}

// ---------------------------------------------------------------------------
// Kernel 2: transpose final-MLP weights to [out][in] and convert to f16 so the
// WMMA B-fragment (one column per lane, 16 consecutive K halves) is a single
// contiguous 32B load.
// ---------------------------------------------------------------------------
__global__ __launch_bounds__(256) void wtr_kernel(
    const float* __restrict__ nw1,  // [192,128]
    const float* __restrict__ nw2,  // [128,128]
    _Float16* __restrict__ w1t,     // [128,192]
    _Float16* __restrict__ w2t)     // [128,128]
{
    int i = blockIdx.x * 256 + threadIdx.x;
    if (i < OUTD * FEAT) {
        int o = i / FEAT, k = i % FEAT;
        w1t[i] = (_Float16)nw1[(size_t)k * OUTD + o];
    }
    int j = i - OUTD * FEAT;
    if (j >= 0 && j < OUTD * OUTD) {
        int o = j >> 7, k = j & 127;
        w2t[j] = (_Float16)nw2[(size_t)k * OUTD + o];
    }
}

// ---------------------------------------------------------------------------
// Kernel 3: message scatter.  64 lanes per edge (one per feature dim):
// coalesced 256B gather of x[src], 3 f32 atomic adds into agg[dst] (L2-resident).
// ---------------------------------------------------------------------------
__global__ __launch_bounds__(256) void scatter_kernel(
    const float* __restrict__ x,    // [N,64]
    const int*   __restrict__ ei,   // [2,E]
    const float* __restrict__ wE,   // [E,4]
    float* __restrict__ agg)        // [N,192]
{
    int gid = blockIdx.x * 256 + threadIdx.x;
    int e = gid >> 6;
    int d = gid & 63;
    if (e >= EE) return;
    int src = ei[e];
    int dst = ei[EE + e];
    float xv = x[(size_t)src * DD + d];
    float w0 = wE[(size_t)e * 4 + 0];
    float w1 = wE[(size_t)e * 4 + 1];
    float w2 = wE[(size_t)e * 4 + 2];
    float* base = agg + (size_t)dst * FEAT + d;
    atomicAdd(base,            xv * w0);
    atomicAdd(base + DD,       xv * w1);
    atomicAdd(base + 2 * DD,   xv * w2);
}

// ---------------------------------------------------------------------------
// Kernel 4: residual + final MLP via WMMA f16->f32.
// One wave (32 threads) per 16-node tile.  feat[16,192] staged in LDS (f16),
// hidden[16,128] staged in LDS (f16), output f32.
// ---------------------------------------------------------------------------
__global__ __launch_bounds__(32) void mlp_kernel(
    const float* __restrict__ agg,  // [N,192]
    const float* __restrict__ x,    // [N,64]
    const float* __restrict__ eps,  // [C]
    const _Float16* __restrict__ w1t, // [128,192] f16 (transposed)
    const float* __restrict__ nb1,  // [128]
    const _Float16* __restrict__ w2t, // [128,128] f16 (transposed)
    const float* __restrict__ nb2,  // [128]
    float* __restrict__ out)        // [N,128]
{
    __shared__ __align__(16) _Float16 ldsA[16][200];  // 192 + pad (400B rows, 16B aligned)
    __shared__ __align__(16) _Float16 ldsH[16][136];  // 128 + pad (272B rows, 16B aligned)

    const int lane = threadIdx.x;
    const int n0 = blockIdx.x * 16;
    const int m  = lane & 15;     // row (A) / column-in-tile (B,C)
    const int hi = lane >> 4;     // half-wave select

    float e0 = 1.0f + eps[0], e1 = 1.0f + eps[1], e2 = 1.0f + eps[2];

    // Stage feat = agg + (1+eps_c)*x into LDS as f16.
    for (int row = 0; row < 16; ++row) {
        const float* arow = agg + (size_t)(n0 + row) * FEAT;
        const float* xrow = x   + (size_t)(n0 + row) * DD;
        for (int col = lane; col < FEAT; col += 32) {
            int c = col >> 6, d = col & 63;
            float ec = (c == 0) ? e0 : ((c == 1) ? e1 : e2);
            ldsA[row][col] = (_Float16)(arow[col] + ec * xrow[d]);
        }
    }
    __syncthreads();

    // -------- Layer 1: hidden[16,128] = leaky(feat[16,192] @ W1 + b1) --------
    v8f acc[8] = {};
#pragma unroll
    for (int kb = 0; kb < 6; ++kb) {                 // K = 192 = 6 * 32
        int koff = kb * 32 + hi * 16;
        // A fragment (16x32, this code's k order, matching B): halves 0-15 = K+16*hi+0..15 (contiguous)
        h8 alo = *(const h8*)&ldsA[m][koff];
        h8 ahi = *(const h8*)&ldsA[m][koff + 8];
        v16h a;
#pragma unroll
        for (int i = 0; i < 8; ++i) { a[i] = alo[i]; a[i + 8] = ahi[i]; }
#pragma unroll
        for (int t = 0; t < 8; ++t) {                // 8 column tiles of 16
            // B fragment (32x16): lane column = t*16+m, 16 contiguous K halves
            const _Float16* bp = w1t + (size_t)(t * 16 + m) * FEAT + kb * 32 + hi * 16;
            v16h b = *(const v16h*)bp;               // 32B aligned -> 2x b128
            acc[t] = __builtin_amdgcn_wmma_f32_16x16x32_f16(
                false, a, false, b, (short)0, acc[t], false, false);
        }
    }
    // Bias + LeakyReLU, store hidden tile to LDS (C/D layout: row = r + hi*8).
#pragma unroll
    for (int t = 0; t < 8; ++t) {
        int col = t * 16 + m;
        float bias = nb1[col];
#pragma unroll
        for (int r = 0; r < 8; ++r) {
            float v = acc[t][r] + bias;
            v = (v >= 0.0f) ? v : NEG_SLOPE * v;
            ldsH[r + hi * 8][col] = (_Float16)v;
        }
    }
    __syncthreads();

    // -------- Layer 2: out[16,128] = hidden[16,128] @ W2 + b2 --------
    v8f acc2[8] = {};
#pragma unroll
    for (int kb = 0; kb < 4; ++kb) {                 // K = 128 = 4 * 32
        int koff = kb * 32 + hi * 16;
        h8 alo = *(const h8*)&ldsH[m][koff];
        h8 ahi = *(const h8*)&ldsH[m][koff + 8];
        v16h a;
#pragma unroll
        for (int i = 0; i < 8; ++i) { a[i] = alo[i]; a[i + 8] = ahi[i]; }
#pragma unroll
        for (int t = 0; t < 8; ++t) {
            const _Float16* bp = w2t + (size_t)(t * 16 + m) * OUTD + kb * 32 + hi * 16;
            v16h b = *(const v16h*)bp;
            acc2[t] = __builtin_amdgcn_wmma_f32_16x16x32_f16(
                false, a, false, b, (short)0, acc2[t], false, false);
        }
    }
#pragma unroll
    for (int t = 0; t < 8; ++t) {
        int col = t * 16 + m;
        float bias = nb2[col];
#pragma unroll
        for (int r = 0; r < 8; ++r) {
            out[(size_t)(n0 + r + hi * 8) * OUTD + col] = acc2[t][r] + bias;
        }
    }
}

// ---------------------------------------------------------------------------
extern "C" void kernel_launch(void* const* d_in, const int* in_sizes, int n_in,
                              void* d_out, int out_size, void* d_ws, size_t ws_size,
                              hipStream_t stream) {
    const float* x   = (const float*)d_in[0];
    const int*   ei  = (const int*)  d_in[1];   // [2,E] int32
    const float* ea  = (const float*)d_in[2];
    const float* w1  = (const float*)d_in[3];
    const float* b1  = (const float*)d_in[4];
    const float* w2  = (const float*)d_in[5];
    const float* b2  = (const float*)d_in[6];
    const float* eps = (const float*)d_in[7];
    const float* nw1 = (const float*)d_in[8];
    const float* nb1 = (const float*)d_in[9];
    const float* nw2 = (const float*)d_in[10];
    const float* nb2 = (const float*)d_in[11];
    float* out = (float*)d_out;

    // Workspace layout
    float*    agg = (float*)d_ws;                        // N*192 f32 (38.4 MB)
    float*    wE  = agg + (size_t)NN * FEAT;             // E*4 f32  (12.8 MB)
    _Float16* w1t = (_Float16*)(wE + (size_t)EE * 4);    // 128*192 f16
    _Float16* w2t = w1t + OUTD * FEAT;                   // 128*128 f16

    // agg must be zero at the start of every (graph-replayed) launch.
    hipMemsetAsync(agg, 0, (size_t)NN * FEAT * sizeof(float), stream);

    edgew_kernel<<<(EE + 255) / 256, 256, 0, stream>>>(ea, w1, b1, w2, b2, wE);
    wtr_kernel<<<(OUTD * FEAT + OUTD * OUTD + 255) / 256, 256, 0, stream>>>(nw1, nw2, w1t, w2t);
    scatter_kernel<<<(size_t)EE * 64 / 256, 256, 0, stream>>>(x, ei, wE, agg);
    mlp_kernel<<<NN / 16, 32, 0, stream>>>(agg, x, eps, w1t, nb1, w2t, nb2, out);
}
